// MultiHeadedAttention_66529043415281
// MI455X (gfx1250) — compile-verified, matrix operand order checked by reading, not run on hardware
//
#include <hip/hip_runtime.h>
#include <hip/hip_bf16.h>

// CDNA5 / gfx1250 WMMA types
typedef __attribute__((ext_vector_type(16))) __bf16 v16bf;
typedef __attribute__((ext_vector_type(8)))  __bf16 v8bf;
typedef __attribute__((ext_vector_type(8)))  float  v8f;

// ---------------------------------------------------------------------------
// One-shot f32 -> bf16 conversion (8 elements / thread, b128 in, b128 out).
// ---------------------------------------------------------------------------
__global__ __launch_bounds__(256) void f32_to_bf16(const float* __restrict__ in,
                                                   __bf16* __restrict__ out, int n8) {
  const int i = blockIdx.x * blockDim.x + threadIdx.x;
  if (i >= n8) return;
  const float4* p = reinterpret_cast<const float4*>(in) + (size_t)i * 2;
  float4 a = p[0], b = p[1];
  v8bf o;
  o[0] = (__bf16)a.x; o[1] = (__bf16)a.y; o[2] = (__bf16)a.z; o[3] = (__bf16)a.w;
  o[4] = (__bf16)b.x; o[5] = (__bf16)b.y; o[6] = (__bf16)b.z; o[7] = (__bf16)b.w;
  reinterpret_cast<v8bf*>(out)[i] = o;
}

// ---------------------------------------------------------------------------
// One 16x32 bf16 WMMA fragment from a row-major bf16 matrix (global).
// ISA layout: lane l holds row (l&15);
//   elements 0..7  = K  kk + 8*(l>>4) + 0..7      (16B load)
//   elements 8..15 = K  kk + 16 + 8*(l>>4) + 0..7 (16B load)
// ---------------------------------------------------------------------------
__device__ __forceinline__ v16bf load_frag(const __bf16* __restrict__ base,
                                           int ld, int row0, int kk, int lane) {
  const __bf16* p = base + (size_t)(row0 + (lane & 15)) * ld + (kk + 8 * (lane >> 4));
  v8bf lo = *reinterpret_cast<const v8bf*>(p);
  v8bf hi = *reinterpret_cast<const v8bf*>(p + 16);
  v16bf f;
#pragma unroll
  for (int i = 0; i < 8; ++i) { f[i] = lo[i]; f[8 + i] = hi[i]; }
  return f;
}

// Same fragment, read from an LDS-resident 64x32 bf16 panel (row-major).
__device__ __forceinline__ v16bf load_frag_lds(const __bf16* panel, int row0, int lane) {
  const __bf16* p = panel + ((row0 + (lane & 15)) << 5) + 8 * (lane >> 4);
  v8bf lo = *reinterpret_cast<const v8bf*>(p);       // ds_load_b128
  v8bf hi = *reinterpret_cast<const v8bf*>(p + 16);  // ds_load_b128
  v16bf f;
#pragma unroll
  for (int i = 0; i < 8; ++i) { f[i] = lo[i]; f[8 + i] = hi[i]; }
  return f;
}

// ---------------------------------------------------------------------------
// Out[M,N] = X[M,K] * W[N,K]^T + bias[N]   (X, W pre-converted bf16)
//
// Block = 64x256 tile, 8 waves; wave = 64x32 sub-tile (4 A-frags x 2 B-frags
// -> 8 WMMA per K-step of 32). The 64-row A panel is shared by all 8 waves:
// staged per K-step with gfx1250 GLOBAL_LOAD_ASYNC_TO_LDS_B128 (256 threads
// x 16B = 4KB panel), TRIPLE-buffered (async issued 2 steps ahead, after the
// barrier) so only ONE barrier per K-step is needed, ASYNCcnt-tracked.
// B fragments are unique per wave: direct global b128 loads issued before the
// async wait so they overlap the barrier.
// ---------------------------------------------------------------------------
__global__ __launch_bounds__(256) void gemm_bias_wmma(
    const __bf16* __restrict__ X, const __bf16* __restrict__ W,
    const float* __restrict__ bias, float* __restrict__ Out,
    int M, int N, int K) {
  __shared__ __align__(16) __bf16 apanel[3][64 * 32];   // 3 x 4KB

  const int tid  = threadIdx.x;
  const int lane = tid & 31;
  const int wave = tid >> 5;
  const int ntb  = N >> 8;                          // 256-wide block columns
  const int tm   = (blockIdx.x / ntb) << 6;         // 64 rows per block
  const int tn   = ((blockIdx.x % ntb) << 8) + (wave << 5);

  // async-copy mapping: thread t copies 16B of row (t>>2), K-chunk (t&3)*8
  const int crow = tid >> 2;
  const int cchk = (tid & 3) << 3;
  const uint64_t g0   = (uint64_t)(uintptr_t)(X + (size_t)(tm + crow) * K + cchk);
  const uint32_t lds0 = (uint32_t)(uintptr_t)&apanel[0][(crow << 5) + cchk];
  const __bf16*  ap0  = &apanel[0][0];

  // issue one 16B async copy: panel byte-offset `boff`, K offset `kk` elements
#define ISSUE_ASYNC(boff, kk)                                                   \
  asm volatile("global_load_async_to_lds_b128 %0, %1, off"                      \
               :: "v"(lds0 + (uint32_t)(boff)),                                 \
                  "v"(g0 + (uint64_t)((kk) * 2)) : "memory")

  v8f c[4][2];
#pragma unroll
  for (int i = 0; i < 4; ++i) { c[i][0] = (v8f){}; c[i][1] = (v8f){}; }

  // one K-step of matrix math: A frags from LDS panel at byte-offset bo
  auto mma_step = [&](uint32_t bo, v16bf b0, v16bf b1) {
    const __bf16* ab = ap0 + (bo >> 1);
    v16bf a0 = load_frag_lds(ab, 0,  lane);
    v16bf a1 = load_frag_lds(ab, 16, lane);
    v16bf a2 = load_frag_lds(ab, 32, lane);
    v16bf a3 = load_frag_lds(ab, 48, lane);
    c[0][0] = __builtin_amdgcn_wmma_f32_16x16x32_bf16(false, a0, false, b0, (short)0, c[0][0], false, false);
    c[0][1] = __builtin_amdgcn_wmma_f32_16x16x32_bf16(false, a0, false, b1, (short)0, c[0][1], false, false);
    c[1][0] = __builtin_amdgcn_wmma_f32_16x16x32_bf16(false, a1, false, b0, (short)0, c[1][0], false, false);
    c[1][1] = __builtin_amdgcn_wmma_f32_16x16x32_bf16(false, a1, false, b1, (short)0, c[1][1], false, false);
    c[2][0] = __builtin_amdgcn_wmma_f32_16x16x32_bf16(false, a2, false, b0, (short)0, c[2][0], false, false);
    c[2][1] = __builtin_amdgcn_wmma_f32_16x16x32_bf16(false, a2, false, b1, (short)0, c[2][1], false, false);
    c[3][0] = __builtin_amdgcn_wmma_f32_16x16x32_bf16(false, a3, false, b0, (short)0, c[3][0], false, false);
    c[3][1] = __builtin_amdgcn_wmma_f32_16x16x32_bf16(false, a3, false, b1, (short)0, c[3][1], false, false);
  };

  // prologue: two panels in flight
  ISSUE_ASYNC(0, 0);
  ISSUE_ASYNC(4096, 32);

  uint32_t bo = 0;                          // current read panel (bytes: 0/4096/8192)
  int kk = 0;
  // steady state: branch-free body, one barrier per K-step
  for (; kk < K - 64; kk += 32) {
    v16bf b0 = load_frag(W, K, tn,      kk, lane);
    v16bf b1 = load_frag(W, K, tn + 16, kk, lane);
    asm volatile("s_wait_asynccnt 0x1" ::: "memory");    // oldest (current panel) done
    __syncthreads();                                     // panel visible; prev reads done
    const uint32_t bo2 = (bo >= 4096) ? bo - 4096 : 8192;  // (bo + 8192) % 12288
    ISSUE_ASYNC(bo2, kk + 64);                           // overwrites panel read at kk-32
    mma_step(bo, b0, b1);
    bo = (bo == 8192) ? 0 : bo + 4096;
  }
  {   // kk == K-64: nothing left to issue
    v16bf b0 = load_frag(W, K, tn,      kk, lane);
    v16bf b1 = load_frag(W, K, tn + 16, kk, lane);
    asm volatile("s_wait_asynccnt 0x1" ::: "memory");
    __syncthreads();
    mma_step(bo, b0, b1);
    bo = (bo == 8192) ? 0 : bo + 4096;
    kk += 32;
  }
  {   // kk == K-32: last panel
    v16bf b0 = load_frag(W, K, tn,      kk, lane);
    v16bf b1 = load_frag(W, K, tn + 16, kk, lane);
    asm volatile("s_wait_asynccnt 0x0" ::: "memory");
    __syncthreads();
    mma_step(bo, b0, b1);
  }
#undef ISSUE_ASYNC

  // C/D layout: VGPR e -> M = e + 8*(lane>>4), N = lane&15
  const int mr  = tm + ((lane >> 4) << 3);
  const int nc  = tn + (lane & 15);
  const float bn0 = bias[nc];
  const float bn1 = bias[nc + 16];
  float* o = Out + (size_t)mr * N + nc;                  // 32-bit offsets below
#pragma unroll
  for (int i = 0; i < 4; ++i) {
    float* oi = o + i * 16 * N;
#pragma unroll
    for (int e = 0; e < 8; ++e) {
      oi[e * N]      = c[i][0][e] + bn0;
      oi[e * N + 16] = c[i][1][e] + bn1;
    }
  }
}

// ---------------------------------------------------------------------------
// Per-token outer-product softmax attention.
// Block = 64 threads = one (b,t,h) head; thread i owns score row i.
// Row i of scores is q_i * (k/8), so rowmax = max(q_i*kmax, q_i*kmin)
// with kmax/kmin shared per head -> single exp pass per row.
// ---------------------------------------------------------------------------
__global__ __launch_bounds__(64) void attn_token_softmax(
    const float* __restrict__ Q, const float* __restrict__ K,
    const float* __restrict__ V, float* __restrict__ H) {
  __shared__ float ks[64], vs[64], rmx[64], rmn[64];
  const int i    = threadIdx.x;
  const int row  = blockIdx.x >> 4;            // token index (b*T + t)
  const int head = blockIdx.x & 15;
  const size_t base = (size_t)row * 1024 + head * 64;

  const float kv = K[base + i];
  ks[i] = kv;  vs[i] = V[base + i];
  rmx[i] = kv; rmn[i] = kv;
  __syncthreads();
#pragma unroll
  for (int off = 32; off > 0; off >>= 1) {
    if (i < off) {
      rmx[i] = fmaxf(rmx[i], rmx[i + off]);
      rmn[i] = fminf(rmn[i], rmn[i + off]);
    }
    __syncthreads();
  }

  const float qs = Q[base + i] * 0.125f;       // 1/sqrt(DK=64)
  const float m  = fmaxf(qs * rmx[0], qs * rmn[0]);
  float den = 0.f, num = 0.f;
#pragma unroll
  for (int j = 0; j < 64; ++j) {
    const float e = __expf(qs * ks[j] - m);    // v_exp_f32
    den += e;
    num += e * vs[j];
  }
  H[base + i] = num / den;
}

// ---------------------------------------------------------------------------
extern "C" void kernel_launch(void* const* d_in, const int* in_sizes, int n_in,
                              void* d_out, int out_size, void* d_ws, size_t ws_size,
                              hipStream_t stream) {
  const float* query = (const float*)d_in[0];
  const float* key   = (const float*)d_in[1];
  const float* value = (const float*)d_in[2];
  const float* Wq    = (const float*)d_in[3];   // [H*DK, DU] row-major
  const float* bq    = (const float*)d_in[4];
  const float* Wk    = (const float*)d_in[5];
  const float* bk    = (const float*)d_in[6];
  const float* Wv    = (const float*)d_in[7];
  const float* bv    = (const float*)d_in[8];
  const float* Wo    = (const float*)d_in[9];   // [DU, DU]
  const float* bo    = (const float*)d_in[10];
  float* out = (float*)d_out;

  const int DU = 1024, H = 16;
  const int M  = in_sizes[0] / DU;              // B*T = 2048
  const size_t MD = (size_t)M * DU;
  const size_t WD = (size_t)DU * DU;

  // Workspace: f32 Q,K,V,H (32 MB) then bf16 copies (24 MB)
  float* Qb = (float*)d_ws;
  float* Kb = Qb + MD;
  float* Vb = Kb + MD;
  float* Hb = Vb + MD;
  __bf16* Xqb = (__bf16*)(Hb + MD);
  __bf16* Xkb = Xqb + MD;
  __bf16* Xvb = Xkb + MD;
  __bf16* Hbf = Xvb + MD;
  __bf16* Wqb = Hbf + MD;
  __bf16* Wkb = Wqb + WD;
  __bf16* Wvb = Wkb + WD;
  __bf16* Wob = Wvb + WD;

  // --- one-shot f32 -> bf16 conversions ---
  const int cvtX = (int)(MD / 8), cvtW = (int)(WD / 8);
  const int gX = (cvtX + 255) / 256, gW = (cvtW + 255) / 256;
  f32_to_bf16<<<gX, 256, 0, stream>>>(query, Xqb, cvtX);
  f32_to_bf16<<<gX, 256, 0, stream>>>(key,   Xkb, cvtX);
  f32_to_bf16<<<gX, 256, 0, stream>>>(value, Xvb, cvtX);
  f32_to_bf16<<<gW, 256, 0, stream>>>(Wq, Wqb, cvtW);
  f32_to_bf16<<<gW, 256, 0, stream>>>(Wk, Wkb, cvtW);
  f32_to_bf16<<<gW, 256, 0, stream>>>(Wv, Wvb, cvtW);
  f32_to_bf16<<<gW, 256, 0, stream>>>(Wo, Wob, cvtW);

  // --- projections: 64x256 block tiles (8 waves), async-staged A panel ---
  const int nblocks = (M / 64) * (DU / 256);    // 128 blocks
  gemm_bias_wmma<<<nblocks, 256, 0, stream>>>(Xqb, Wqb, bq, Qb, M, DU, DU);
  gemm_bias_wmma<<<nblocks, 256, 0, stream>>>(Xkb, Wkb, bk, Kb, M, DU, DU);
  gemm_bias_wmma<<<nblocks, 256, 0, stream>>>(Xvb, Wvb, bv, Vb, M, DU, DU);

  // --- per-token softmax attention ---
  attn_token_softmax<<<M * H, 64, 0, stream>>>(Qb, Kb, Vb, Hb);

  // --- output projection ---
  f32_to_bf16<<<gX, 256, 0, stream>>>(Hb, Hbf, cvtX);
  gemm_bias_wmma<<<nblocks, 256, 0, stream>>>(Hbf, Wob, bo, out, M, DU, DU);
}
